// HiddenStateODEFunc_80041010528505
// MI455X (gfx1250) — compile-verified, matrix operand order checked by reading, not run on hardware
//
#include <hip/hip_runtime.h>
#include <math.h>

// ---------------------------------------------------------------------------
// Fused  Linear -> LayerNorm -> exact GELU -> Linear  for MI455X (gfx1250)
// B=16384, H=1024, D=4096, fp32 in/out; f16 WMMA compute with f32 accumulation
// BM=32 rows per workgroup: every weight fragment feeds 2 WMMAs (register
// reuse), halving L2 weight traffic vs a 16-row tile. z-tile lives in 256 KB
// of the WGP's 320 KB LDS.
// ---------------------------------------------------------------------------

typedef __attribute__((ext_vector_type(16))) _Float16 v16h;
typedef __attribute__((ext_vector_type(8)))  _Float16 v8h;
typedef __attribute__((ext_vector_type(8)))  float    v8f;
typedef __attribute__((ext_vector_type(4)))  float    v4f;

#define B_SZ 16384
#define H_SZ 1024
#define D_SZ 4096
#define BM   32
#define LN_EPS 1e-5f

// Shared memory: z-tile (32 x 4096 f16) + reduction scratch
#define SMEM_Z_BYTES   (BM * D_SZ * 2)                       // 262144
#define SMEM_BYTES     (SMEM_Z_BYTES + (256 + 256 + BM + BM) * 4)

// ---------------------------------------------------------------------------
// Load a 16x32 (MxK) f16 WMMA fragment. Works for both A fragments (rows) and
// B fragments (columns of a pre-transposed weight matrix): "m" walks the
// 16-wide tile dimension, K is contiguous in memory with stride `ldk`.
// Per-lane layout (CDNA5 ISA 7.12.2, 16-bit A 16x32):
//   lanes 0-15 : m = lane,    halves 0..7 = K k0..k0+7, halves 8..15 = K+16..23
//   lanes 16-31: m = lane-16, halves 0..7 = K k0+8..15, halves 8..15 = K+24..31
// => two contiguous 16-byte loads per lane.
// ---------------------------------------------------------------------------
__device__ __forceinline__ v16h load_frag16x32(const _Float16* __restrict__ base,
                                               int ldk, int k0, int lane) {
  const int m  = lane & 15;
  const int hi = (lane >> 4) & 1;
  const _Float16* p = base + (size_t)m * ldk + (k0 + hi * 8);
  v8h lo = *(const v8h*)(p);
  v8h hh = *(const v8h*)(p + 16);
  v16h r;
#pragma unroll
  for (int i = 0; i < 8; ++i) { r[i] = lo[i]; r[i + 8] = hh[i]; }
  return r;
}

// ---------------------------------------------------------------------------
// Prep kernel 1: fp32 -> f16 bulk convert (hidden_state), 8 elems/thread
// ---------------------------------------------------------------------------
__global__ void cvt_f32_to_f16_x8(const float* __restrict__ in,
                                  _Float16* __restrict__ out) {
  const size_t i = ((size_t)blockIdx.x * blockDim.x + threadIdx.x) * 8;
  v4f a = *(const v4f*)(in + i);
  v4f b = *(const v4f*)(in + i + 4);
  v8h o;
#pragma unroll
  for (int j = 0; j < 4; ++j) { o[j] = (_Float16)a[j]; o[j + 4] = (_Float16)b[j]; }
  *(v8h*)(out + i) = o;
}

// ---------------------------------------------------------------------------
// Prep kernel 2: transpose + convert weights:  in[rows][cols] -> out[cols][rows]
// (makes every WMMA B-fragment K-contiguous)
// ---------------------------------------------------------------------------
__global__ void transpose_cvt_f16(const float* __restrict__ in,
                                  _Float16* __restrict__ out,
                                  int rows, int cols) {
  const size_t i = (size_t)blockIdx.x * blockDim.x + threadIdx.x;
  const int c = (int)(i % (size_t)cols);
  const int r = (int)(i / (size_t)cols);
  out[(size_t)c * rows + r] = (_Float16)in[i];
}

// ---------------------------------------------------------------------------
// Fused main kernel: one 32-row stripe of the batch per workgroup.
// 256 threads = 8 wave32's.
// ---------------------------------------------------------------------------
__global__ void __launch_bounds__(256)
fused_mlp_ln_gelu(const _Float16* __restrict__ hs16,   // [B][H]
                  const _Float16* __restrict__ w1t,    // [D][H]  (W1 transposed)
                  const _Float16* __restrict__ w2t,    // [H][D]  (W2 transposed)
                  const float* __restrict__ b1,        // [D]
                  const float* __restrict__ gamma,     // [D]
                  const float* __restrict__ beta,      // [D]
                  const float* __restrict__ b2,        // [H]
                  float* __restrict__ out)             // [B][H]
{
  extern __shared__ char smem[];
  _Float16* zbuf   = (_Float16*)smem;                        // 32 x 4096 f16
  float*    psum   = (float*)(smem + SMEM_Z_BYTES);          // 256
  float*    psq    = psum + 256;                             // 256
  float*    s_mean = psq + 256;                              // 32
  float*    s_rstd = s_mean + BM;                            // 32

  const int tid  = threadIdx.x;
  const int wid  = tid >> 5;
  const int lane = tid & 31;
  const int row0 = blockIdx.x * BM;

  // C/D tile layout (CDNA5 16x16 f32): VGPR r, lane L -> M = r + 8*(L>>4), N = L&15
  const int mb  = (lane >> 4) * 8;
  const int ncl = lane & 15;

  const _Float16* aBase = hs16 + (size_t)row0 * H_SZ;

  // ---------------- Phase 1: z = hs @ W1 + b1  ->  zbuf (f16) ---------------
#pragma unroll 1
  for (int nc = 0; nc < 8; ++nc) {
    const int n0 = wid * 512 + nc * 64;   // this wave's 64-column slab
    v8f acc[2][4] = {};
#pragma unroll 2
    for (int k0 = 0; k0 < H_SZ; k0 += 32) {
      v16h a0 = load_frag16x32(aBase, H_SZ, k0, lane);
      v16h a1 = load_frag16x32(aBase + 16 * H_SZ, H_SZ, k0, lane);
#pragma unroll
      for (int t = 0; t < 4; ++t) {
        v16h b = load_frag16x32(w1t + (size_t)(n0 + t * 16) * H_SZ, H_SZ, k0, lane);
        acc[0][t] = __builtin_amdgcn_wmma_f32_16x16x32_f16(
            false, a0, false, b, (short)0, acc[0][t], false, false);
        acc[1][t] = __builtin_amdgcn_wmma_f32_16x16x32_f16(
            false, a1, false, b, (short)0, acc[1][t], false, false);
      }
    }
#pragma unroll
    for (int t = 0; t < 4; ++t) {
      const int col = n0 + t * 16 + ncl;
      const float bb = b1[col];
#pragma unroll
      for (int m = 0; m < 2; ++m)
#pragma unroll
        for (int r = 0; r < 8; ++r)
          zbuf[(m * 16 + mb + r) * D_SZ + col] = (_Float16)(acc[m][t][r] + bb);
    }
  }
  __syncthreads();

  // ---------------- Phase 2: per-row mean / rstd over D ---------------------
  {
    const int row = tid >> 3;        // 8 threads per row, 32 rows
    const int sub = tid & 7;
    float s = 0.f, sq = 0.f;
    const _Float16* zp = zbuf + row * D_SZ + sub * 512;
#pragma unroll 8
    for (int i = 0; i < 512; ++i) { const float x = (float)zp[i]; s += x; sq += x * x; }
    psum[tid] = s; psq[tid] = sq;
  }
  __syncthreads();
  if (tid < BM) {
    float s = 0.f, sq = 0.f;
#pragma unroll
    for (int i = 0; i < 8; ++i) { s += psum[tid * 8 + i]; sq += psq[tid * 8 + i]; }
    const float mu  = s * (1.0f / D_SZ);
    const float var = sq * (1.0f / D_SZ) - mu * mu;
    s_mean[tid] = mu;
    s_rstd[tid] = rsqrtf(var + LN_EPS);
  }
  __syncthreads();

  // ---------------- Phase 3: LayerNorm + exact GELU in place ----------------
#pragma unroll 1
  for (int e = tid; e < BM * D_SZ; e += 256) {
    const int row = e >> 12;
    const int col = e & (D_SZ - 1);
    const float x  = (float)zbuf[e];
    const float xn = (x - s_mean[row]) * s_rstd[row] * gamma[col] + beta[col];
    const float g  = 0.5f * xn * (1.0f + erff(xn * 0.70710678118654752f));
    zbuf[e] = (_Float16)g;
  }
  __syncthreads();

  // ---------------- Phase 4: out = act @ W2 + b2 ----------------------------
  {
    const int n0 = wid * 128;  // this wave's 128 output columns
    v8f acc[2][8] = {};
#pragma unroll 1
    for (int k0 = 0; k0 < D_SZ; k0 += 32) {
      v16h a0 = load_frag16x32(zbuf, D_SZ, k0, lane);              // from LDS
      v16h a1 = load_frag16x32(zbuf + 16 * D_SZ, D_SZ, k0, lane);  // from LDS
#pragma unroll
      for (int t = 0; t < 8; ++t) {
        v16h b = load_frag16x32(w2t + (size_t)(n0 + t * 16) * D_SZ, D_SZ, k0, lane);
        acc[0][t] = __builtin_amdgcn_wmma_f32_16x16x32_f16(
            false, a0, false, b, (short)0, acc[0][t], false, false);
        acc[1][t] = __builtin_amdgcn_wmma_f32_16x16x32_f16(
            false, a1, false, b, (short)0, acc[1][t], false, false);
      }
    }
#pragma unroll
    for (int t = 0; t < 8; ++t) {
      const int col = n0 + t * 16 + ncl;
      const float bb = b2[col];
#pragma unroll
      for (int m = 0; m < 2; ++m)
#pragma unroll
        for (int r = 0; r < 8; ++r)
          out[(size_t)(row0 + m * 16 + mb + r) * H_SZ + col] = acc[m][t][r] + bb;
    }
  }
}

// ---------------------------------------------------------------------------
// Host launcher
// d_in: [0]_time [1]hidden_state [2]W1 [3]b1 [4]ln_gamma [5]ln_beta [6]W2 [7]b2
// ---------------------------------------------------------------------------
extern "C" void kernel_launch(void* const* d_in, const int* in_sizes, int n_in,
                              void* d_out, int out_size, void* d_ws, size_t ws_size,
                              hipStream_t stream) {
  const float* hs    = (const float*)d_in[1];
  const float* W1    = (const float*)d_in[2];
  const float* b1    = (const float*)d_in[3];
  const float* gamma = (const float*)d_in[4];
  const float* beta  = (const float*)d_in[5];
  const float* W2    = (const float*)d_in[6];
  const float* b2    = (const float*)d_in[7];
  float* out = (float*)d_out;

  // Workspace layout (f16): hidden [B*H] | W1^T [D*H] | W2^T [H*D]  = 48 MB
  _Float16* hs16 = (_Float16*)d_ws;
  _Float16* w1t  = hs16 + (size_t)B_SZ * H_SZ;
  _Float16* w2t  = w1t + (size_t)D_SZ * H_SZ;

  // One-time precision conversion / weight transposes (graph-capture safe)
  cvt_f32_to_f16_x8<<<(B_SZ * H_SZ) / (256 * 8), 256, 0, stream>>>(hs, hs16);
  transpose_cvt_f16<<<(H_SZ * D_SZ) / 256, 256, 0, stream>>>(W1, w1t, H_SZ, D_SZ);
  transpose_cvt_f16<<<(D_SZ * H_SZ) / 256, 256, 0, stream>>>(W2, w2t, D_SZ, H_SZ);

  // ~258 KB of dynamic LDS per workgroup (CDNA5 WGP has 320 KB)
  (void)hipFuncSetAttribute((const void*)fused_mlp_ln_gelu,
                            hipFuncAttributeMaxDynamicSharedMemorySize, SMEM_BYTES);
  fused_mlp_ln_gelu<<<B_SZ / BM, 256, SMEM_BYTES, stream>>>(
      hs16, w1t, w2t, b1, gamma, beta, b2, out);
}
